// get_model_29678224016156
// MI455X (gfx1250) — matrix operand order repaired and checked by reading, not compiled, not run on hardware
//
#include <hip/hip_runtime.h>
#include <hip/hip_bf16.h>

#define BB 8  // batch

typedef __attribute__((ext_vector_type(16))) _Float16 v16h;
typedef __attribute__((ext_vector_type(8)))  float    v8f;

static inline int ru16(int x){ return (x+15)&~15; }
static inline int ru32(int x){ return (x+31)&~31; }
static inline int cdiv(int a,int b){ return (a+b-1)/b; }

// ---------------------------------------------------------------------------
// Split (B,6,N) -> pts (B,N,3), norm (B,N,3)
__global__ void k_split_xyz(const float* __restrict__ xyz, float* __restrict__ pts,
                            float* __restrict__ nrm, int N)
{
    int t = blockIdx.x*blockDim.x + threadIdx.x;
    if (t >= BB*N) return;
    int b = t / N, n = t % N;
    const float* src = xyz + (size_t)b*6*N + n;
    float* p = pts + ((size_t)b*N + n)*3;
    float* q = nrm + ((size_t)b*N + n)*3;
    p[0]=src[0];     p[1]=src[(size_t)N];   p[2]=src[(size_t)2*N];
    q[0]=src[(size_t)3*N]; q[1]=src[(size_t)4*N]; q[2]=src[(size_t)5*N];
}

// ---------------------------------------------------------------------------
// Farthest point sampling. One block (256 thr) per batch; coords+dist in regs.
template<int N>
__global__ void __launch_bounds__(256) k_fps(const float* __restrict__ pts,
                                             int* __restrict__ out, int npoint)
{
    constexpr int NPT = N/256;
    int b = blockIdx.x, tid = threadIdx.x;
    const float* P = pts + (size_t)b*N*3;
    float px[NPT], py[NPT], pz[NPT], dist[NPT];
    #pragma unroll
    for (int i=0;i<NPT;++i){
        int j = tid + i*256;
        px[i]=P[j*3]; py[i]=P[j*3+1]; pz[i]=P[j*3+2];
        dist[i]=1e10f;
    }
    __shared__ float sv[256];
    __shared__ int   si[256];
    int far = 0;
    for (int it=0; it<npoint; ++it){
        if (tid==0) out[b*npoint+it] = far;
        float cx=P[far*3], cy=P[far*3+1], cz=P[far*3+2];
        float best=-1.0f; int bi=N;
        #pragma unroll
        for (int i=0;i<NPT;++i){
            float dx=px[i]-cx, dy=py[i]-cy, dz=pz[i]-cz;
            float d = dx*dx+dy*dy+dz*dz;
            dist[i] = fminf(dist[i], d);
            if (dist[i] > best){ best = dist[i]; bi = tid + i*256; }
        }
        sv[tid]=best; si[tid]=bi;
        __syncthreads();
        for (int s=128;s>0;s>>=1){
            if (tid<s){
                if (sv[tid+s] > sv[tid] ||
                    (sv[tid+s]==sv[tid] && si[tid+s]<si[tid])){
                    sv[tid]=sv[tid+s]; si[tid]=si[tid+s];
                }
            }
            __syncthreads();
        }
        far = si[0];
        __syncthreads();
    }
}

// ---------------------------------------------------------------------------
__global__ void k_gather_xyz(const float* __restrict__ pts, const int* __restrict__ fidx,
                             float* __restrict__ out, int N, int S)
{
    int t = blockIdx.x*blockDim.x + threadIdx.x;
    if (t >= BB*S) return;
    int b = t / S, s = t % S;
    int j = fidx[b*S+s];
    const float* p = pts + ((size_t)b*N + j)*3;
    float* o = out + ((size_t)b*S + s)*3;
    o[0]=p[0]; o[1]=p[1]; o[2]=p[2];
}

// ---------------------------------------------------------------------------
// Ball query: first K indices (ascending) within radius, pad with first.
__global__ void k_query_ball(const float* __restrict__ pts, const float* __restrict__ nxyz,
                             int* __restrict__ gidx, int N, int S, float r2, int K)
{
    int t = blockIdx.x*blockDim.x + threadIdx.x;
    if (t >= BB*S) return;
    int b = t / S, s = t % S;
    const float* c = nxyz + ((size_t)b*S + s)*3;
    float cx=c[0], cy=c[1], cz=c[2];
    int* g = gidx + ((size_t)b*S + s)*K;
    const float* P = pts + (size_t)b*N*3;
    int cnt = 0;
    for (int j=0; j<N; ++j){
        float dx=P[j*3]-cx, dy=P[j*3+1]-cy, dz=P[j*3+2]-cz;
        float d = dx*dx+dy*dy+dz*dz;
        if (d <= r2){ g[cnt++] = j; if (cnt==K) break; }
    }
    int f = (cnt>0) ? g[0] : 0;
    for (; cnt<K; ++cnt) g[cnt] = f;
}

// ---------------------------------------------------------------------------
// Build A rows: [feat(gidx), xyz(gidx)-new_xyz], zero-padded to Kpad, f16.
__global__ void k_gather_group(const float* __restrict__ feat, int featC,
                               const float* __restrict__ pts, const float* __restrict__ nxyz,
                               const int* __restrict__ gidx, _Float16* __restrict__ A,
                               int N, int S, int s0, int Schunk, int K, int Kpad, int cin)
{
    int r = blockIdx.x*blockDim.x + threadIdx.x;
    int Mc = BB*Schunk*K;
    if (r >= Mc) return;
    int k = r % K; int u = r / K; int sl = u % Schunk; int b = u / Schunk;
    int s = s0 + sl;
    int j = gidx[((size_t)(b*S + s))*K + k];
    _Float16* Ar = A + (size_t)r * Kpad;
    const float* F = feat + ((size_t)b*N + j)*(size_t)featC;
    for (int c=0; c<featC; ++c) Ar[c] = (_Float16)F[c];
    const float* pj = pts + ((size_t)b*N + j)*3;
    const float* nx = nxyz + ((size_t)b*S + s)*3;
    Ar[featC+0] = (_Float16)(pj[0]-nx[0]);
    Ar[featC+1] = (_Float16)(pj[1]-nx[1]);
    Ar[featC+2] = (_Float16)(pj[2]-nx[2]);
    for (int c=cin; c<Kpad; ++c) Ar[c] = (_Float16)0.f;
}

// ---------------------------------------------------------------------------
// Transpose+convert weights (Cout,Cin) f32 -> Wt (Kpad,Npad) f16, zero-padded.
__global__ void k_prep_w(const float* __restrict__ w, _Float16* __restrict__ wt,
                         int Cout, int Cin, int Npad, int Kpad)
{
    int t = blockIdx.x*blockDim.x + threadIdx.x;
    if (t >= Kpad*Npad) return;
    int n = t % Npad; int k = t / Npad;
    wt[t] = (n < Cout && k < Cin) ? (_Float16)w[(size_t)n*Cin + k] : (_Float16)0.f;
}

// ---------------------------------------------------------------------------
// WMMA GEMM + BN + ReLU epilogue. One wave per 32x32 output block (2x2 tiles of
// 16x16), reusing each A fragment across 2 N-tiles and each B fragment across
// 2 M-tiles: 4 b128 A loads + 32 strided 16-bit B loads -> 4 WMMAs per k-step.
// A: (M,Kpad) f16 row-major, Wt: (Kpad,Npad) f16 row-major (= W transposed).
union AFrag { v16h v; float4 f4[2]; };

__device__ __forceinline__ void epilogue_store(
    v8f acc, int tm, int tn, int lane, int Npad,
    const float* __restrict__ bias, const float* __restrict__ g,
    const float* __restrict__ beta, const float* __restrict__ mean,
    const float* __restrict__ var,
    int Cout, int do_bn, int do_relu,
    _Float16* __restrict__ outh, float* __restrict__ outf)
{
    int n = tn*16 + (lane & 15);
    int mbase = tm*16 + ((lane >> 4) ? 8 : 0);
    float scale, shift;
    if (n < Cout){
        if (do_bn){
            float rs = rsqrtf(var[n] + 1e-5f);
            scale = g[n]*rs;
            shift = beta[n] + (bias[n]-mean[n])*scale;
        } else { scale = 1.f; shift = bias[n]; }
    } else { scale = 0.f; shift = 0.f; }
    #pragma unroll
    for (int v=0; v<8; ++v){
        float y = acc[v]*scale + shift;
        if (do_relu) y = fmaxf(y, 0.f);
        size_t o = (size_t)(mbase + v)*Npad + n;
        if (outf) outf[o] = y;
        else      outh[o] = (_Float16)y;
    }
}

__global__ void __launch_bounds__(256)
k_gemm_wmma(const _Float16* __restrict__ A, const _Float16* __restrict__ Wt,
            int M, int Npad, int Kpad,
            const float* __restrict__ bias, const float* __restrict__ g,
            const float* __restrict__ beta, const float* __restrict__ mean,
            const float* __restrict__ var,
            int Cout, int do_bn, int do_relu,
            _Float16* __restrict__ outh, float* __restrict__ outf)
{
    int lane = threadIdx.x & 31;
    int wave = threadIdx.x >> 5;
    int tiles_m = M >> 4, tiles_n = Npad >> 4;
    int tm0 = (blockIdx.x*8 + wave)*2;
    if (tm0 >= tiles_m) return;          // wave-uniform
    int tn0 = blockIdx.y*2;
    // clamp second tiles: edge waves recompute an in-bounds tile and simply
    // don't store it -> inner loop stays branch-free, EXEC stays all-ones.
    int tm1 = (tm0+1 < tiles_m) ? tm0+1 : tm0;
    int tn1 = (tn0+1 < tiles_n) ? tn0+1 : tn0;

    // A fragment per lane: row m = tm*16 + (lane&15),
    // k-halves start at (lane>=16 ? 8 : 0); elems 0..7 -> k+0..7, 8..15 -> k+16..23
    int mr   = lane & 15;
    int aoff = (lane >> 4) * 8;
    const _Float16* Arow0 = A + (size_t)(tm0*16 + mr)*Kpad + aoff;
    const _Float16* Arow1 = A + (size_t)(tm1*16 + mr)*Kpad + aoff;
    // B fragment per lane: column n = tn*16 + (lane&15), k order same as A
    const _Float16* Brow0 = Wt + (size_t)tn0*16 + mr;
    const _Float16* Brow1 = Wt + (size_t)tn1*16 + mr;

    v8f acc00 = {}, acc01 = {}, acc10 = {}, acc11 = {};
    for (int k = 0; k < Kpad; k += 32){
        AFrag a0, a1, b0, b1;
        a0.f4[0] = *(const float4*)(Arow0 + k);
        a0.f4[1] = *(const float4*)(Arow0 + k + 16);
        a1.f4[0] = *(const float4*)(Arow1 + k);
        a1.f4[1] = *(const float4*)(Arow1 + k + 16);
        #pragma unroll
        for (int s = 0; s < 16; ++s){
            size_t kk = (size_t)(k + aoff + (s < 8 ? s : s + 8));
            b0.v[s] = Brow0[kk*Npad];
            b1.v[s] = Brow1[kk*Npad];
        }
        acc00 = __builtin_amdgcn_wmma_f32_16x16x32_f16(
                    false, a0.v, false, b0.v, (short)0, acc00, false, false);
        acc01 = __builtin_amdgcn_wmma_f32_16x16x32_f16(
                    false, a0.v, false, b1.v, (short)0, acc01, false, false);
        acc10 = __builtin_amdgcn_wmma_f32_16x16x32_f16(
                    false, a1.v, false, b0.v, (short)0, acc10, false, false);
        acc11 = __builtin_amdgcn_wmma_f32_16x16x32_f16(
                    false, a1.v, false, b1.v, (short)0, acc11, false, false);
    }

    bool m1ok = (tm0+1 < tiles_m);
    bool n1ok = (tn0+1 < tiles_n);
    epilogue_store(acc00, tm0, tn0, lane, Npad, bias,g,beta,mean,var,
                   Cout, do_bn, do_relu, outh, outf);
    if (n1ok)
        epilogue_store(acc01, tm0, tn1, lane, Npad, bias,g,beta,mean,var,
                       Cout, do_bn, do_relu, outh, outf);
    if (m1ok)
        epilogue_store(acc10, tm1, tn0, lane, Npad, bias,g,beta,mean,var,
                       Cout, do_bn, do_relu, outh, outf);
    if (m1ok && n1ok)
        epilogue_store(acc11, tm1, tn1, lane, Npad, bias,g,beta,mean,var,
                       Cout, do_bn, do_relu, outh, outf);
}

// ---------------------------------------------------------------------------
// Max pool over K rows per (b,s); write into lp[b, s0+s, col0+c].
__global__ void k_maxpool(const _Float16* __restrict__ act, int Npad, int Cout, int K,
                          float* __restrict__ dst, int dstStride, int col0,
                          int Schunk, int s0, int Stotal)
{
    int t = blockIdx.x*blockDim.x + threadIdx.x;
    int total = BB*Schunk*Cout;
    if (t >= total) return;
    int c = t % Cout; int u = t / Cout; int s = u % Schunk; int b = u / Schunk;
    size_t rowbase = ((size_t)(b*Schunk + s))*K;
    float m = -3.4e38f;
    for (int k=0; k<K; ++k)
        m = fmaxf(m, (float)act[(rowbase + k)*Npad + c]);
    dst[((size_t)b*Stotal + (s0+s))*dstStride + col0 + c] = m;
}

// ---------------------------------------------------------------------------
// sa3 input rows: (B*128, 672) = [l2_xyz(3), l2_p(640), pad]
__global__ void k_build_sa3(const float* __restrict__ l2x, const float* __restrict__ l2p,
                            _Float16* __restrict__ A)
{
    int t = blockIdx.x*blockDim.x + threadIdx.x;
    if (t >= 1024*672) return;
    int r = t / 672, c = t % 672;
    float v = 0.f;
    if (c < 3)        v = l2x[(size_t)r*3 + c];
    else if (c < 643) v = l2p[(size_t)r*640 + (c-3)];
    A[t] = (_Float16)v;
}

// ---------------------------------------------------------------------------
// Quantization MLP head: q(8,1) -> 20 relu -> 10 relu
__global__ void k_qpath(const float* __restrict__ q,
                        const float* __restrict__ w1, const float* __restrict__ b1,
                        const float* __restrict__ w2, const float* __restrict__ b2,
                        float* __restrict__ qout)
{
    int b = threadIdx.x;
    if (b >= BB) return;
    float qv = q[b];
    float h[20];
    for (int j=0;j<20;++j) h[j] = fmaxf(w1[j]*qv + b1[j], 0.f);
    for (int i=0;i<10;++i){
        float a = b2[i];
        for (int j=0;j<20;++j) a += w2[i*20+j]*h[j];
        qout[b*10+i] = fmaxf(a, 0.f);
    }
}

// FC input rows: 16 x 1056 = [l3(1024), qout(10), pad]; rows >=8 zero.
__global__ void k_build_fcin(const float* __restrict__ l3, const float* __restrict__ qout,
                             _Float16* __restrict__ A)
{
    int t = blockIdx.x*blockDim.x + threadIdx.x;
    if (t >= 16*1056) return;
    int r = t / 1056, c = t % 1056;
    float v = 0.f;
    if (r < BB){
        if (c < 1024)      v = l3[(size_t)r*1024 + c];
        else if (c < 1034) v = qout[r*10 + (c-1024)];
    }
    A[t] = (_Float16)v;
}

// d_out = [x (8,40), l3^T (8,1024,1) == l3 flat]
__global__ void k_finalize(const float* __restrict__ fcout, const float* __restrict__ l3,
                           float* __restrict__ out)
{
    int t = blockIdx.x*blockDim.x + threadIdx.x;
    if (t >= BB*40 + BB*1024) return;
    if (t < BB*40){
        int b = t/40, c = t%40;
        out[t] = fcout[b*48 + c];
    } else {
        out[t] = l3[t - BB*40];
    }
}

// ---------------------------------------------------------------------------
// Host helpers
static void run_layer(hipStream_t stream,
                      const _Float16* A, int M, int Kpad,
                      const float* w, int cin, int cout,
                      const float* bias, const float* g, const float* beta,
                      const float* mean, const float* var,
                      int do_bn, int do_relu,
                      _Float16* outh, float* outf, _Float16* wt)
{
    int Npad = ru16(cout);
    int tot = Kpad*Npad;
    k_prep_w<<<cdiv(tot,256),256,0,stream>>>(w, wt, cout, cin, Npad, Kpad);
    int tiles_m = M/16, tiles_n = Npad/16;
    dim3 grid(cdiv(tiles_m, 16), cdiv(tiles_n, 2));
    k_gemm_wmma<<<grid,256,0,stream>>>(A, wt, M, Npad, Kpad,
                                       bias, g, beta, mean, var,
                                       cout, do_bn, do_relu, outh, outf);
}

static void run_sa_branch(hipStream_t stream, void* const* d_in,
                          const float* ptsS, const float* featS, int featC,
                          const float* nxyz, int Nst, int Sst, int Schunk,
                          float radius, int K, const int* chans, int paramBase,
                          int* gidx, _Float16* Abuf, _Float16* act0, _Float16* act1,
                          _Float16* wt, float* lp, int lpStride, int col0)
{
    int cin = featC + 3;
    int Kpad_in = ru32(cin);
    k_query_ball<<<cdiv(BB*Sst,256),256,0,stream>>>(ptsS, nxyz, gidx, Nst, Sst,
                                                    radius*radius, K);
    for (int s0 = 0; s0 < Sst; s0 += Schunk){
        int Mc = BB*Schunk*K;
        k_gather_group<<<cdiv(Mc,256),256,0,stream>>>(featS, featC, ptsS, nxyz, gidx,
                                                      Abuf, Nst, Sst, s0, Schunk, K,
                                                      Kpad_in, cin);
        const _Float16* Ain = Abuf;
        int KpadCur = Kpad_in, cinCur = cin;
        _Float16* outs[2] = {act0, act1};
        int lastNpad = 16, lastCout = 16;
        const _Float16* lastAct = act0;
        for (int l=0; l<3; ++l){
            int cout = chans[l];
            int bse = paramBase + l*6;
            run_layer(stream, Ain, Mc, KpadCur,
                      (const float*)d_in[bse+0], cinCur, cout,
                      (const float*)d_in[bse+1], (const float*)d_in[bse+2],
                      (const float*)d_in[bse+3], (const float*)d_in[bse+4],
                      (const float*)d_in[bse+5], 1, 1, outs[l&1], nullptr, wt);
            Ain = outs[l&1]; lastAct = outs[l&1];
            lastNpad = ru16(cout); lastCout = cout;
            KpadCur = lastNpad;    // all channel counts are multiples of 32
            cinCur = cout;
        }
        k_maxpool<<<cdiv(BB*Schunk*lastCout,256),256,0,stream>>>(lastAct, lastNpad,
            lastCout, K, lp, lpStride, col0, Schunk, s0, Sst);
    }
}

// ---------------------------------------------------------------------------
extern "C" void kernel_launch(void* const* d_in, const int* in_sizes, int n_in,
                              void* d_out, int out_size, void* d_ws, size_t ws_size,
                              hipStream_t stream)
{
    (void)in_sizes; (void)n_in; (void)out_size; (void)ws_size;
    const float* xyz   = (const float*)d_in[0];
    const float* quant = (const float*)d_in[1];

    // workspace bump allocator
    char* base = (char*)d_ws; size_t off = 0;
    auto walloc = [&](size_t bytes)->void* {
        void* p = base + off;
        off = (off + bytes + 255) & ~(size_t)255;
        return p;
    };
    float* pts   = (float*)walloc((size_t)BB*4096*3*4);
    float* nrm   = (float*)walloc((size_t)BB*4096*3*4);
    int*   fidx1 = (int*)  walloc((size_t)BB*512*4);
    float* l1x   = (float*)walloc((size_t)BB*512*3*4);
    float* l1p   = (float*)walloc((size_t)BB*512*320*4);
    int*   fidx2 = (int*)  walloc((size_t)BB*128*4);
    float* l2x   = (float*)walloc((size_t)BB*128*3*4);
    float* l2p   = (float*)walloc((size_t)BB*128*640*4);
    float* l3    = (float*)walloc((size_t)BB*1024*4);
    float* qout  = (float*)walloc((size_t)BB*10*4);
    float* fcout = (float*)walloc((size_t)16*48*4);
    int*   gidx  = (int*)  walloc((size_t)BB*512*128*4);
    _Float16* wt   = (_Float16*)walloc((size_t)1056*1024*2);
    _Float16* Abuf = (_Float16*)walloc((size_t)32768*352*2);
    _Float16* act0 = (_Float16*)walloc((size_t)131072*128*2);
    _Float16* act1 = (_Float16*)walloc((size_t)131072*128*2);

    // ---- stage 0: split input
    k_split_xyz<<<cdiv(BB*4096,256),256,0,stream>>>(xyz, pts, nrm, 4096);

    // ---- SA1: FPS(512) + 3 MSG branches, chunked S=128
    k_fps<4096><<<BB,256,0,stream>>>(pts, fidx1, 512);
    k_gather_xyz<<<cdiv(BB*512,256),256,0,stream>>>(pts, fidx1, l1x, 4096, 512);
    {
        const int   K1[3]   = {16, 32, 128};
        const float r1[3]   = {0.1f, 0.2f, 0.4f};
        const int   ch1[3][3] = {{32,32,64},{64,64,128},{64,96,128}};
        int col = 0;
        for (int br=0; br<3; ++br){
            run_sa_branch(stream, d_in, pts, nrm, 3, l1x, 4096, 512, 128,
                          r1[br], K1[br], ch1[br], 2 + br*18,
                          gidx, Abuf, act0, act1, wt, l1p, 320, col);
            col += ch1[br][2];
        }
    }

    // ---- SA2: FPS(128) + 3 MSG branches, chunked S=32
    k_fps<512><<<BB,256,0,stream>>>(l1x, fidx2, 128);
    k_gather_xyz<<<cdiv(BB*128,256),256,0,stream>>>(l1x, fidx2, l2x, 512, 128);
    {
        const int   K2[3]   = {32, 64, 128};
        const float r2[3]   = {0.2f, 0.4f, 0.8f};
        const int   ch2[3][3] = {{64,64,128},{128,128,256},{128,128,256}};
        int col = 0;
        for (int br=0; br<3; ++br){
            run_sa_branch(stream, d_in, l1x, l1p, 320, l2x, 512, 128, 32,
                          r2[br], K2[br], ch2[br], 56 + br*18,
                          gidx, Abuf, act0, act1, wt, l2p, 640, col);
            col += ch2[br][2];
        }
    }

    // ---- SA3 (group-all): rows = B*128, 643ch -> 256 -> 512 -> 1024, max over 128
    k_build_sa3<<<cdiv(1024*672,256),256,0,stream>>>(l2x, l2p, Abuf);
    {
        const int ch3[3] = {256, 512, 1024};
        const _Float16* Ain = Abuf;
        int Kp = 672, cin = 643;
        for (int l=0; l<3; ++l){
            int bse = 110 + l*6;
            run_layer(stream, Ain, 1024, Kp,
                      (const float*)d_in[bse+0], cin, ch3[l],
                      (const float*)d_in[bse+1], (const float*)d_in[bse+2],
                      (const float*)d_in[bse+3], (const float*)d_in[bse+4],
                      (const float*)d_in[bse+5], 1, 1,
                      (l&1) ? act1 : act0, nullptr, wt);
            Ain = (l&1) ? act1 : act0;
            Kp = ru16(ch3[l]); cin = ch3[l];
        }
        // last layer landed in act0 (l=2); pool 128 points per batch
        k_maxpool<<<cdiv(BB*1024,256),256,0,stream>>>(act0, 1024, 1024, 128,
                                                      l3, 1024, 0, 1, 0, 1);
    }

    // ---- FC head
    k_qpath<<<1,32,0,stream>>>(quant,
                               (const float*)d_in[128], (const float*)d_in[129],
                               (const float*)d_in[130], (const float*)d_in[131], qout);
    k_build_fcin<<<cdiv(16*1056,256),256,0,stream>>>(l3, qout, Abuf);
    run_layer(stream, Abuf, 16, 1056,
              (const float*)d_in[132], 1034, 512,
              (const float*)d_in[133], (const float*)d_in[134],
              (const float*)d_in[135], (const float*)d_in[136],
              (const float*)d_in[137], 1, 1, act0, nullptr, wt);
    run_layer(stream, act0, 16, 512,
              (const float*)d_in[138], 512, 256,
              (const float*)d_in[139], (const float*)d_in[140],
              (const float*)d_in[141], (const float*)d_in[142],
              (const float*)d_in[143], 1, 1, act1, nullptr, wt);
    run_layer(stream, act1, 16, 256,
              (const float*)d_in[144], 256, 40,
              (const float*)d_in[145], nullptr, nullptr, nullptr, nullptr,
              0, 0, nullptr, fcout, wt);

    k_finalize<<<cdiv(BB*40 + BB*1024, 256),256,0,stream>>>(fcout, l3, (float*)d_out);
}
